// Tensorproduct_Attention_11690900980363
// MI455X (gfx1250) — matrix operand order repaired and checked by reading, not compiled, not run on hardware
//
#include <hip/hip_runtime.h>

// ---------------- CDNA5 types ----------------
typedef __attribute__((ext_vector_type(16))) __bf16        v16bf;
typedef __attribute__((ext_vector_type(8)))  float         v8f;
typedef __attribute__((ext_vector_type(4)))  unsigned int  v4u;
typedef __attribute__((ext_vector_type(8)))  unsigned int  v8u;

struct B32 { v4u a, b; };

// ---------------- bf16 split helpers ----------------
__device__ __forceinline__ unsigned short bf16_rn(float x) {
  unsigned u = __float_as_uint(x);
  u += 0x7FFFu + ((u >> 16) & 1u);
  return (unsigned short)(u >> 16);
}
__device__ __forceinline__ void bf16_split(float x, unsigned short& hi, unsigned short& lo) {
  hi = bf16_rn(x);
  float hf = __uint_as_float(((unsigned)hi) << 16);
  lo = bf16_rn(x - hf);
}

// B-fragment: 16 contiguous bf16 (32B) per lane from global (weights in ws)
__device__ __forceinline__ v16bf load_bfrag(const unsigned short* p) {
  return __builtin_bit_cast(v16bf, *(const v8u*)p);
}
// A-fragment: two 16B chunks (K+0..7 , K+8..15, contiguous) from LDS
__device__ __forceinline__ v16bf load_afrag(const unsigned short* p) {
  B32 t; t.a = *(const v4u*)p; t.b = *(const v4u*)(p + 8);
  return __builtin_bit_cast(v16bf, t);
}
__device__ __forceinline__ v8f wmma_bf16(v16bf a, v16bf b, v8f c) {
  return __builtin_amdgcn_wmma_f32_16x16x32_bf16(false, a, false, b, (short)0, c, false, false);
}

// ---------------- ws layout (ushort elements) ----------------
// wqkv_hi [3 mat][3 l][128 v][128 u] @0, wqkv_lo @147456
// ls_hi [256][128] @294912, ls_lo @327680
// lvs_hi [128][128] @360448, lvs_lo @376832   (total 393216 ushort = 768KB)

__global__ __launch_bounds__(256) void tpa_prep(
    const float* __restrict__ Wq, const float* __restrict__ Wk, const float* __restrict__ Wv,
    const float* __restrict__ lsw, const float* __restrict__ lvsw,
    unsigned short* __restrict__ wsb)
{
  int i = blockIdx.x * 256 + threadIdx.x;  // 196608 total elements
  float x; unsigned hi_at, lo_at;
  if (i < 147456) {
    x = (i < 49152) ? Wq[i] : (i < 98304) ? Wk[i - 49152] : Wv[i - 98304];
    hi_at = i;            lo_at = 147456 + i;
  } else if (i < 180224) {
    int j = i - 147456;   x = lsw[j];
    hi_at = 294912 + j;   lo_at = 327680 + j;
  } else {
    int j = i - 180224;   x = lvsw[j];
    hi_at = 360448 + j;   lo_at = 376832 + j;
  }
  unsigned short h, l; bf16_split(x, h, l);
  wsb[hi_at] = h; wsb[lo_at] = l;
}

// ---------------- fused main kernel: 32 rows / workgroup ----------------
// LDS map (bytes):
//  aT_hi [2][9][16][128] u16 @0        (73728)
//  aT_lo                 @73728        (73728)
//  aS_hi [2][16][128]    @147456       (8192)   s-matrix frags
//  aS_lo                 @155648       (8192)
//  aX_hi [2][16][128]    @163840       (8192)   S-input frags
//  aX_lo                 @172032       (8192)
//  logits [2][16][256] f32 @180224     (32768)
//  ostage [16][128][9] f32 @212992     (73728)
//  w2l    [3][128] f32    @286720      (1536)   => 288256 total (<320KB WGP LDS)
__global__ __launch_bounds__(256) void tpa_main(
    const float* __restrict__ T, const float* __restrict__ S,
    const float* __restrict__ w2, const float* __restrict__ ls_b,
    const float* __restrict__ lvs_b, const unsigned short* __restrict__ wsb,
    float* __restrict__ out_tu, float* __restrict__ out2)
{
  extern __shared__ char smem[];
  unsigned short* aT_hi = (unsigned short*)(smem);
  unsigned short* aT_lo = (unsigned short*)(smem + 73728);
  unsigned short* aS_hi = (unsigned short*)(smem + 147456);
  unsigned short* aS_lo = (unsigned short*)(smem + 155648);
  unsigned short* aX_hi = (unsigned short*)(smem + 163840);
  unsigned short* aX_lo = (unsigned short*)(smem + 172032);
  float*          logit = (float*)(smem + 180224);
  float*          ostage= (float*)(smem + 212992);
  float*          w2l   = (float*)(smem + 286720);

  const int tid  = threadIdx.x;
  const int lane = tid & 31, wv = tid >> 5;
  const int m16  = lane & 15;
  const int koffB = (lane & 16) ? 16 : 0;   // B frag: K+16 for upper half-wave
  const int koffA = (lane & 16) ? 16 : 0;   // A frag: K+16 for upper half-wave
  const int rofs  = (lane >> 4) << 3;       // D-tile: upper half holds rows +8
  const int colw  = wv * 16 + m16;          // wave-owned output column
  const size_t n0 = (size_t)blockIdx.x * 32;

  // ---- Phase 0: stage T, S (bf16 hi/lo split), w2 into LDS ----
  {
    const float* gT = T + n0 * 1152;
    for (int e = tid; e < 36864; e += 256) {
      float x = __builtin_nontemporal_load(gT + e);
      int nl = e / 1152, r = e % 1152, u = r / 9, d = r % 9;
      int a = (((nl >> 4) * 9 + d) * 16 + (nl & 15)) * 128 + u;
      unsigned short h, l; bf16_split(x, h, l);
      aT_hi[a] = h; aT_lo[a] = l;
    }
    const float* gS = S + n0 * 128;
    for (int e = tid; e < 4096; e += 256) {
      float x = __builtin_nontemporal_load(gS + e);
      unsigned short h, l; bf16_split(x, h, l);
      aX_hi[e] = h; aX_lo[e] = l;
    }
    for (int e = tid; e < 384; e += 256) w2l[e] = w2[e];
  }
  __syncthreads();

  // ---- Phase 1: q,k GEMMs (bf16x3 split WMMA), accumulate s in registers ----
  v8f sreg0 = {}; v8f sreg1 = {};
  for (int l = 0; l < 3; ++l) {
    const int cnt = 2 * l + 1, db = l * l;
    v16bf Bq[4][2], Bk[4][2];                 // hoisted weight frags (128 VGPRs)
    const int base_q = ((0 * 3 + l) * 128 + colw) * 128 + koffB;
    const int base_k = ((1 * 3 + l) * 128 + colw) * 128 + koffB;
#pragma unroll
    for (int kc = 0; kc < 4; ++kc) {
      Bq[kc][0] = load_bfrag(wsb + base_q + kc * 32);
      Bq[kc][1] = load_bfrag(wsb + 147456 + base_q + kc * 32);
      Bk[kc][0] = load_bfrag(wsb + base_k + kc * 32);
      Bk[kc][1] = load_bfrag(wsb + 147456 + base_k + kc * 32);
    }
    // C1_l / F  (folds both 1/sqrt(F) factors of q and k)
    float scl = 0.57735026919f * rsqrtf((float)(2 * l + 1)) * (1.0f / 128.0f);
#pragma unroll
    for (int s = 0; s < 2; ++s) {
      v8f& sreg = s ? sreg1 : sreg0;
      for (int dl = 0; dl < cnt; ++dl) {
        int rowbase = ((s * 9 + db + dl) * 16 + m16) * 128 + koffA;
        v8f qa = {}, ka = {};
#pragma unroll
        for (int kc = 0; kc < 4; ++kc) {
          v16bf ah = load_afrag(aT_hi + rowbase + kc * 32);
          v16bf al = load_afrag(aT_lo + rowbase + kc * 32);
          qa = wmma_bf16(al, Bq[kc][0], qa);
          qa = wmma_bf16(ah, Bq[kc][1], qa);
          qa = wmma_bf16(ah, Bq[kc][0], qa);
          ka = wmma_bf16(al, Bk[kc][0], ka);
          ka = wmma_bf16(ah, Bk[kc][1], ka);
          ka = wmma_bf16(ah, Bk[kc][0], ka);
        }
#pragma unroll
        for (int j = 0; j < 8; ++j) sreg[j] += scl * qa[j] * ka[j];
      }
    }
  }

  // ---- Phase 2: s -> bf16 hi/lo A-fragments in LDS ----
#pragma unroll
  for (int s = 0; s < 2; ++s) {
    v8f sreg = s ? sreg1 : sreg0;
#pragma unroll
    for (int j = 0; j < 8; ++j) {
      unsigned short h, l; bf16_split(sreg[j], h, l);
      int a = (s * 16 + j + rofs) * 128 + colw;
      aS_hi[a] = h; aS_lo[a] = l;
    }
  }
  __syncthreads();

  // ---- Phase 3: logits = s @ ls_w^T + ls_b  (16x256, K=128) ----
#pragma unroll
  for (int s = 0; s < 2; ++s) {
#pragma unroll
    for (int jt = 0; jt < 2; ++jt) {
      int colL = (wv * 2 + jt) * 16 + m16;   // 0..255
      v8f acc = {};
      int abase = (s * 16 + m16) * 128 + koffA;
      int bbase = colL * 128 + koffB;
#pragma unroll
      for (int kc = 0; kc < 4; ++kc) {
        v16bf ah = load_afrag(aS_hi + abase + kc * 32);
        v16bf al = load_afrag(aS_lo + abase + kc * 32);
        v16bf bh = load_bfrag(wsb + 294912 + bbase + kc * 32);
        v16bf bl = load_bfrag(wsb + 327680 + bbase + kc * 32);
        acc = wmma_bf16(al, bh, acc);
        acc = wmma_bf16(ah, bl, acc);
        acc = wmma_bf16(ah, bh, acc);
      }
      float bias = ls_b[colL];
#pragma unroll
      for (int j = 0; j < 8; ++j)
        logit[(s * 16 + j + rofs) * 256 + colL] = acc[j] + bias;
    }
  }
  __syncthreads();

  // ---- Phase 4: softmax over 256 per row (8 threads / row, shfl reduce) ----
  {
    int r = tid >> 3, part = tid & 7;
    float* row = logit + r * 256;
    float mx = -3.4e38f;
    for (int c = part * 32; c < part * 32 + 32; ++c) mx = fmaxf(mx, row[c]);
    mx = fmaxf(mx, __shfl_xor(mx, 1));
    mx = fmaxf(mx, __shfl_xor(mx, 2));
    mx = fmaxf(mx, __shfl_xor(mx, 4));
    float sum = 0.f;
    for (int c = part * 32; c < part * 32 + 32; ++c) {
      float e = __expf(row[c] - mx);
      row[c] = e; sum += e;
    }
    sum += __shfl_xor(sum, 1);
    sum += __shfl_xor(sum, 2);
    sum += __shfl_xor(sum, 4);
    float inv = 1.0f / sum;
    for (int c = part * 32; c < part * 32 + 32; ++c) row[c] *= inv;
  }
  __syncthreads();

  // ---- Phase 5: out2 = su * silu(S @ lvs_w^T + lvs_b) ----
  {
    float bias = lvs_b[colw];
#pragma unroll
    for (int s = 0; s < 2; ++s) {
      v8f acc = {};
      int abase = (s * 16 + m16) * 128 + koffA;
      int bbase = colw * 128 + koffB;
#pragma unroll
      for (int kc = 0; kc < 4; ++kc) {
        v16bf ah = load_afrag(aX_hi + abase + kc * 32);
        v16bf al = load_afrag(aX_lo + abase + kc * 32);
        v16bf bh = load_bfrag(wsb + 360448 + bbase + kc * 32);
        v16bf bl = load_bfrag(wsb + 376832 + bbase + kc * 32);
        acc = wmma_bf16(al, bh, acc);
        acc = wmma_bf16(ah, bl, acc);
        acc = wmma_bf16(ah, bh, acc);
      }
#pragma unroll
      for (int j = 0; j < 8; ++j) {
        int row = j + rofs;
        float y  = acc[j] + bias;
        float sg = 1.0f / (1.0f + __expf(-y));
        float su = logit[(s * 16 + row) * 256 + colw];
        __builtin_nontemporal_store(su * y * sg,
            out2 + (n0 + s * 16 + row) * 128 + colw);
      }
    }
  }

  // ---- Phase 6: v GEMM, scale by sd*w2/sqrt(F), stage tu in LDS, coalesced out ----
  for (int s = 0; s < 2; ++s) {
    for (int l = 0; l < 3; ++l) {
      const int cnt = 2 * l + 1, db = l * l;
      v16bf Bv[4][2];
      int base_v = ((2 * 3 + l) * 128 + colw) * 128 + koffB;
#pragma unroll
      for (int kc = 0; kc < 4; ++kc) {
        Bv[kc][0] = load_bfrag(wsb + base_v + kc * 32);
        Bv[kc][1] = load_bfrag(wsb + 147456 + base_v + kc * 32);
      }
      float wf = w2l[l * 128 + colw] * 0.08838834764831845f;   // w2 * 1/sqrt(F)
      for (int dl = 0; dl < cnt; ++dl) {
        int d = db + dl;
        int rowbase = ((s * 9 + d) * 16 + m16) * 128 + koffA;
        v8f acc = {};
#pragma unroll
        for (int kc = 0; kc < 4; ++kc) {
          v16bf ah = load_afrag(aT_hi + rowbase + kc * 32);
          v16bf al = load_afrag(aT_lo + rowbase + kc * 32);
          acc = wmma_bf16(al, Bv[kc][0], acc);
          acc = wmma_bf16(ah, Bv[kc][1], acc);
          acc = wmma_bf16(ah, Bv[kc][0], acc);
        }
#pragma unroll
        for (int j = 0; j < 8; ++j) {
          int row = j + rofs;
          float sd = logit[(s * 16 + row) * 256 + 128 + colw];
          ostage[(row * 128 + colw) * 9 + d] = acc[j] * sd * wf;
        }
      }
    }
    __syncthreads();
    const v4u* src = (const v4u*)ostage;                 // 73728B = 4608 x 16B
    v4u* dst = (v4u*)(out_tu + (n0 + s * 16) * 1152);
    for (int i = tid; i < 4608; i += 256)
      __builtin_nontemporal_store(src[i], dst + i);
    __syncthreads();
  }
}

// ---------------- launcher ----------------
extern "C" void kernel_launch(void* const* d_in, const int* in_sizes, int n_in,
                              void* d_out, int out_size, void* d_ws, size_t ws_size,
                              hipStream_t stream) {
  (void)in_sizes; (void)n_in; (void)out_size; (void)ws_size;
  const float* T    = (const float*)d_in[0];
  const float* S    = (const float*)d_in[1];
  const float* Wq   = (const float*)d_in[2];
  const float* Wk   = (const float*)d_in[3];
  const float* Wv   = (const float*)d_in[4];
  const float* w2   = (const float*)d_in[5];
  const float* lsw  = (const float*)d_in[6];
  const float* lsb  = (const float*)d_in[7];
  const float* lvsw = (const float*)d_in[8];
  const float* lvsb = (const float*)d_in[9];
  float* out_tu = (float*)d_out;
  float* out2   = out_tu + (size_t)49152 * 128 * 9;
  unsigned short* wsb = (unsigned short*)d_ws;

  tpa_prep<<<768, 256, 0, stream>>>(Wq, Wk, Wv, lsw, lvsw, wsb);
  tpa_main<<<1536, 256, 288256, stream>>>(T, S, w2, lsb, lvsb, wsb, out_tu, out2);
}
